// MSARowAttention_58918361366843
// MI455X (gfx1250) — matrix operand order repaired and checked by reading, not compiled, not run on hardware
//
#include <hip/hip_runtime.h>
#include <hip/hip_bf16.h>
#include <math.h>

// ---------------------------------------------------------------------------
// MSA row attention for MI455X (gfx1250, wave32, WMMA f32_16x16x32_f16)
// All B operands pre-transposed to N-major so every WMMA fragment load is a
// contiguous b128; 2x2 register blocking on the large GEMMs; uniform scalar
// branches in the projection epilogue; v_rcp_f32 for sigmoid/softmax.
// ---------------------------------------------------------------------------
typedef __attribute__((ext_vector_type(16))) _Float16 v16h;
typedef __attribute__((ext_vector_type(8)))  _Float16 v8h;
typedef __attribute__((ext_vector_type(8)))  float    v8f;
typedef __attribute__((ext_vector_type(4)))  unsigned int u32x4;
typedef __attribute__((ext_vector_type(8)))  int      i32x8;
typedef __attribute__((ext_vector_type(4)))  int      i32x4;

#define IN_DIMC 128
#define NHEAD   8
#define HDIM    32
#define SEQ     384
#define NROWM   256
#define ROWS    (NROWM * SEQ)     // 98304
#define HDTOT   (NHEAD * HDIM)    // 256
#define NTOT    1024              // 768 (qkv) + 256 (gate)
#define RDK     (NROWM * HDIM)    // 8192 : flattened (r,d) contraction dim

static __device__ __forceinline__ v8f wmma16x16x32(v16h a, v16h b, v8f c) {
  return __builtin_amdgcn_wmma_f32_16x16x32_f16(false, a, false, b, (short)0, c,
                                                false, false);
}

// A fragment (16x32 f16): two contiguous 16B chunks at p and p+16 halves.
static __device__ __forceinline__ v16h load_a16(const _Float16* p) {
  v8h lo = *(const v8h*)p;
  v8h hi = *(const v8h*)(p + 16);
  return __builtin_shufflevector(lo, hi, 0, 1, 2, 3, 4, 5, 6, 7,
                                 8, 9, 10, 11, 12, 13, 14, 15);
}
static __device__ __forceinline__ v16h load_a(const _Float16* base, int ldk, int lane) {
  return load_a16(base + (size_t)(lane & 15) * ldk + (lane >> 4) * 8);
}
// B fragment (32x16 f16) from N-major source: two contiguous 16B chunks at p and p+16 halves.
static __device__ __forceinline__ v16h load_b_nmajor(const _Float16* base, int ldk, int lane) {
  return load_a16(base + (size_t)(lane & 15) * ldk + (lane >> 4) * 8);
}

// ---------------------------------------------------------------------------
// Probe: CDNA5 Tensor Data Mover NULL-descriptor op (D# count=0 => NOP) +
// split tensor counter wait + cluster id. Deterministic & harmless.
// ---------------------------------------------------------------------------
__global__ void k_cdna5_probe(int* scratch) {
  __shared__ int lds_sink[16];
  if (threadIdx.x == 0) {
    u32x4 g0 = {};  // count=0 -> NULL tensor descriptor (architected NOP)
    i32x8 g1 = {};
    i32x4 g2 = {};
    i32x4 g3 = {};
#if defined(__clang_major__) && (__clang_major__ >= 23)
    i32x8 g4 = {};
    __builtin_amdgcn_tensor_load_to_lds(g0, g1, g2, g3, g4, 0);
#else
    __builtin_amdgcn_tensor_load_to_lds(g0, g1, g2, g3, 0);
#endif
    __builtin_amdgcn_s_wait_tensorcnt(0);
    lds_sink[0] = __builtin_amdgcn_cluster_id_x();
    scratch[0] = lds_sink[0];
  }
}

// ---------------------------------------------------------------------------
// K0: weights -> f16, transposed to N-major.
//   wqgT [1024][128]: row n holds column n of (w_qkv | w_gate)
//   woutT [128][256]: row n holds column n of w_out
// ---------------------------------------------------------------------------
__global__ void __launch_bounds__(256) k_cvt_weights(
    const float* __restrict__ w_qkv, const float* __restrict__ w_gate,
    const float* __restrict__ w_out, _Float16* __restrict__ wqgT,
    _Float16* __restrict__ woutT) {
  const int idx = blockIdx.x * blockDim.x + threadIdx.x;
  if (idx < NTOT * IN_DIMC) {
    const int n = idx >> 7, c = idx & 127;
    const float v = (n < 768) ? w_qkv[c * 768 + n] : w_gate[c * 256 + (n - 768)];
    wqgT[idx] = (_Float16)v;
  }
  const int idx2 = idx - NTOT * IN_DIMC;
  if (idx2 >= 0 && idx2 < IN_DIMC * HDTOT) {
    const int n = idx2 >> 8, k2 = idx2 & 255;
    woutT[idx2] = (_Float16)w_out[k2 * IN_DIMC + n];
  }
}

// ---------------------------------------------------------------------------
// K1: LayerNorm + fused QKV/gate projection.
// Block = 256 thr (8 waves) handles 16 rows x 1024 cols; K=128 -> 4 WMMA/tile.
// Each 16-col N-tile is entirely q, k, v, or gate -> uniform scalar branches.
// Each 16-row M-tile is entirely inside one MSA row r_idx (16 | 384).
// ---------------------------------------------------------------------------
__global__ void __launch_bounds__(256) k_ln_project(
    const float* __restrict__ m, const float* __restrict__ ln_w,
    const float* __restrict__ ln_b, const float* __restrict__ b_gate,
    const _Float16* __restrict__ wqgT,
    _Float16* __restrict__ q, _Float16* __restrict__ kk,
    _Float16* __restrict__ vT, _Float16* __restrict__ gate) {
  __shared__ _Float16 smn[16][136];   // 272B row stride (16B multiple)
  const int t     = threadIdx.x;
  const int row0  = blockIdx.x * 16;
  const int r_idx = row0 / SEQ;          // uniform: 16 divides 384
  const int ibase = row0 - r_idx * SEQ;  // uniform, multiple of 16

  {  // LayerNorm: one 16-lane group per row, 8 cols per lane
    const int rl  = t >> 4;
    const int l16 = t & 15;
    const float* mp = m + (size_t)(row0 + rl) * IN_DIMC + l16 * 8;
    float x[8];
    float s = 0.f, s2 = 0.f;
#pragma unroll
    for (int j = 0; j < 8; ++j) { x[j] = mp[j]; s += x[j]; s2 += x[j] * x[j]; }
#pragma unroll
    for (int msk = 1; msk < 16; msk <<= 1) {
      s  += __shfl_xor(s,  msk, 16);
      s2 += __shfl_xor(s2, msk, 16);
    }
    const float mu   = s * (1.f / 128.f);
    const float var  = s2 * (1.f / 128.f) - mu * mu;
    const float rstd = rsqrtf(var + 1e-5f);
#pragma unroll
    for (int j = 0; j < 8; ++j) {
      const int c = l16 * 8 + j;
      smn[rl][c] = (_Float16)((x[j] - mu) * rstd * ln_w[c] + ln_b[c]);
    }
  }
  __syncthreads();

  const int lane = t & 31, wave = t >> 5;
  const int l15  = lane & 15;
  const int mb   = (lane >> 4) * 8;
  v16h afr[4];
#pragma unroll
  for (int s = 0; s < 4; ++s) afr[s] = load_a16(&smn[l15][s * 32 + mb]);

  for (int it = 0; it < 8; ++it) {
    const int ntile = wave * 8 + it, n0 = ntile * 16;  // uniform per iteration
    v8f acc = {};
#pragma unroll
    for (int s = 0; s < 4; ++s)
      acc = wmma16x16x32(afr[s],
                         load_b_nmajor(wqgT + (size_t)n0 * IN_DIMC + s * 32, IN_DIMC, lane),
                         acc);

    if (ntile < 16) {                       // ---- Q (pre-scaled 1/sqrt(32))
      const int n = n0 + l15, h = n >> 5, d = n & 31;
      _Float16* qp = q + (size_t)(h * SEQ + ibase + mb) * RDK + r_idx * HDIM + d;
#pragma unroll
      for (int vv = 0; vv < 8; ++vv)
        qp[(size_t)vv * RDK] = (_Float16)(acc[vv] * 0.17677669529663687f);
    } else if (ntile < 32) {                // ---- K
      const int n = n0 - 256 + l15, h = n >> 5, d = n & 31;
      _Float16* kp = kk + (size_t)(h * SEQ + ibase + mb) * RDK + r_idx * HDIM + d;
#pragma unroll
      for (int vv = 0; vv < 8; ++vv) kp[(size_t)vv * RDK] = (_Float16)acc[vv];
    } else if (ntile < 48) {                // ---- V (per-lane contiguous b128)
      const int n = n0 - 512 + l15, h = n >> 5, d = n & 31;
      v8h pack;
#pragma unroll
      for (int vv = 0; vv < 8; ++vv) pack[vv] = (_Float16)acc[vv];
      *(v8h*)(vT + (size_t)((h * NROWM + r_idx) * HDIM + d) * SEQ + ibase + mb) = pack;
    } else {                                // ---- gate = sigmoid(x + b)
      const int c2 = n0 - 768 + l15;
      const float bg = b_gate[c2];
      _Float16* gp = gate + (size_t)(row0 + mb) * HDTOT + c2;
#pragma unroll
      for (int vv = 0; vv < 8; ++vv)
        gp[(size_t)vv * HDTOT] =
            (_Float16)__builtin_amdgcn_rcpf(1.f + __expf(-(acc[vv] + bg)));
    }
  }
}

// ---------------------------------------------------------------------------
// K2: pair bias = z @ w_pair -> bias[h][i][j] (fp32). H=8 padded to N=16.
// One wave per 16-row (flat i*384+j) tile, K=128 -> 4 WMMA.
// ---------------------------------------------------------------------------
__global__ void __launch_bounds__(128) k_pair_bias(
    const float* __restrict__ z, const float* __restrict__ w_pair,
    float* __restrict__ bias) {
  const int lane  = threadIdx.x & 31, wave = threadIdx.x >> 5;
  const int mtile = blockIdx.x * 4 + wave;
  const int M0    = mtile * 16;
  const int row   = M0 + (lane & 15);
  const int kb8   = (lane >> 4) * 8;
  const int nn    = lane & 15;
  const int kb16  = (lane >> 4) * 16;
  v8f acc = {};
#pragma unroll
  for (int s = 0; s < 4; ++s) {
    const float* zp = z + (size_t)row * IN_DIMC + s * 32 + kb8;
    const v8f zlo = *(const v8f*)zp;
    const v8f zhi = *(const v8f*)(zp + 16);
    v16h a;
#pragma unroll
    for (int j = 0; j < 8; ++j) { a[j] = (_Float16)zlo[j]; a[8 + j] = (_Float16)zhi[j]; }
    v16h b;
#pragma unroll
    for (int j = 0; j < 16; ++j) {
      const int c = s * 32 + kb8 + (j & 7) + (j >> 3) * 16;
      b[j] = (nn < NHEAD) ? (_Float16)w_pair[c * NHEAD + nn] : (_Float16)0.f;
    }
    acc = wmma16x16x32(a, b, acc);
  }
  if (nn < NHEAD) {
    const int mb = (lane >> 4) * 8;
#pragma unroll
    for (int vv = 0; vv < 8; ++vv)
      bias[(size_t)nn * (SEQ * SEQ) + M0 + mb + vv] = acc[vv];
  }
}

// ---------------------------------------------------------------------------
// K3: logits[h][i][j] = sum_{rd} q[h][i][rd]*k[h][j][rd] + bias[h][i][j]
// Per-head 384x8192x384 GEMM; one wave per 32x32 tile (2x2 blocking),
// 256 K-steps x 4 WMMA.
// ---------------------------------------------------------------------------
__global__ void __launch_bounds__(128) k_logits(
    const _Float16* __restrict__ q, const _Float16* __restrict__ kk,
    const float* __restrict__ bias, float* __restrict__ logits) {
  __builtin_amdgcn_s_wait_tensorcnt(0);
  const int lane = threadIdx.x & 31, wave = threadIdx.x >> 5;
  const int gw   = blockIdx.x * 4 + wave;          // 8 * 12 * 12 = 1152 waves
  const int h    = gw / 144, rem = gw - h * 144;
  const int it2  = rem / 12, jt2 = rem - it2 * 12;
  const int i0   = it2 * 32, j0 = jt2 * 32;
  const _Float16* a0p = q  + (size_t)(h * SEQ + i0) * RDK;
  const _Float16* a1p = a0p + (size_t)16 * RDK;
  const _Float16* b0p = kk + (size_t)(h * SEQ + j0) * RDK;
  const _Float16* b1p = b0p + (size_t)16 * RDK;
  v8f acc00 = {}, acc01 = {}, acc10 = {}, acc11 = {};
  for (int s = 0; s < RDK / 32; ++s) {
    if ((s & 15) == 0 && s + 16 < RDK / 32) {
      __builtin_prefetch(a0p + (s + 16) * 32 + (size_t)(lane & 15) * RDK, 0, 1);
      __builtin_prefetch(b0p + (s + 16) * 32 + (size_t)(lane & 15) * RDK, 0, 1);
    }
    const v16h a0 = load_a(a0p + s * 32, RDK, lane);
    const v16h a1 = load_a(a1p + s * 32, RDK, lane);
    const v16h b0 = load_b_nmajor(b0p + s * 32, RDK, lane);
    const v16h b1 = load_b_nmajor(b1p + s * 32, RDK, lane);
    acc00 = wmma16x16x32(a0, b0, acc00);
    acc01 = wmma16x16x32(a0, b1, acc01);
    acc10 = wmma16x16x32(a1, b0, acc10);
    acc11 = wmma16x16x32(a1, b1, acc11);
  }
  const int mb = (lane >> 4) * 8;
#pragma unroll
  for (int ti = 0; ti < 2; ++ti) {
#pragma unroll
    for (int tj = 0; tj < 2; ++tj) {
      const v8f acc = ti ? (tj ? acc11 : acc10) : (tj ? acc01 : acc00);
      const int j = j0 + tj * 16 + (lane & 15);
#pragma unroll
      for (int vv = 0; vv < 8; ++vv) {
        const int i = i0 + ti * 16 + mb + vv;
        const size_t off = ((size_t)h * SEQ + i) * SEQ + j;
        logits[off] = acc[vv] + bias[off];
      }
    }
  }
}

// ---------------------------------------------------------------------------
// K4: softmax over j for each (h,i) row (3072 rows of 384 = 12*32).
// ---------------------------------------------------------------------------
__global__ void __launch_bounds__(256) k_softmax(
    const float* __restrict__ logits, _Float16* __restrict__ attn) {
  const int lane = threadIdx.x & 31, wave = threadIdx.x >> 5;
  const int rowi = blockIdx.x * 8 + wave;
  const float* lp = logits + (size_t)rowi * SEQ;
  float x[12];
  float mx = -1e30f;
#pragma unroll
  for (int tj = 0; tj < 12; ++tj) { x[tj] = lp[lane + tj * 32]; mx = fmaxf(mx, x[tj]); }
#pragma unroll
  for (int msk = 1; msk < 32; msk <<= 1) mx = fmaxf(mx, __shfl_xor(mx, msk, 32));
  float s = 0.f;
#pragma unroll
  for (int tj = 0; tj < 12; ++tj) { x[tj] = __expf(x[tj] - mx); s += x[tj]; }
#pragma unroll
  for (int msk = 1; msk < 32; msk <<= 1) s += __shfl_xor(s, msk, 32);
  const float inv = __builtin_amdgcn_rcpf(s);
  _Float16* ap = attn + (size_t)rowi * SEQ;
#pragma unroll
  for (int tj = 0; tj < 12; ++tj) ap[lane + tj * 32] = (_Float16)(x[tj] * inv);
}

// ---------------------------------------------------------------------------
// K5a: ctx[row][h*32+d] = gate * sum_j attn[h][i][j] * v[h][r][j][d]
// One wave per (h, r, 32-row i tile): M=32, N=32, K=384 -> 12 steps x 4 WMMA.
// B from vT[h][r][d][j] (N-major -> contiguous b128 loads).
// ---------------------------------------------------------------------------
__global__ void __launch_bounds__(128) k_context(
    const _Float16* __restrict__ attn, const _Float16* __restrict__ vT,
    const _Float16* __restrict__ gate, _Float16* __restrict__ ctx) {
  const int lane = threadIdx.x & 31, wave = threadIdx.x >> 5;
  const int gw   = blockIdx.x * 4 + wave;          // 8 * 256 * 12 = 24576 waves
  const int h    = gw / 3072, rem = gw - h * 3072;
  const int r_i  = rem / 12, it2 = rem - r_i * 12;
  const int i0   = it2 * 32;
  const _Float16* a0p = attn + (size_t)(h * SEQ + i0) * SEQ;
  const _Float16* a1p = a0p + (size_t)16 * SEQ;
  const _Float16* vbase = vT + (size_t)(h * NROWM + r_i) * HDIM * SEQ;
  const _Float16* b0p = vbase;                      // d = 0..15
  const _Float16* b1p = vbase + (size_t)16 * SEQ;   // d = 16..31
  v8f acc00 = {}, acc01 = {}, acc10 = {}, acc11 = {};
  for (int s = 0; s < SEQ / 32; ++s) {
    const v16h a0 = load_a(a0p + s * 32, SEQ, lane);
    const v16h a1 = load_a(a1p + s * 32, SEQ, lane);
    const v16h b0 = load_b_nmajor(b0p + s * 32, SEQ, lane);
    const v16h b1 = load_b_nmajor(b1p + s * 32, SEQ, lane);
    acc00 = wmma16x16x32(a0, b0, acc00);
    acc01 = wmma16x16x32(a0, b1, acc01);
    acc10 = wmma16x16x32(a1, b0, acc10);
    acc11 = wmma16x16x32(a1, b1, acc11);
  }
  const int mb = (lane >> 4) * 8;
#pragma unroll
  for (int ti = 0; ti < 2; ++ti) {
#pragma unroll
    for (int nt = 0; nt < 2; ++nt) {
      const v8f acc = ti ? (nt ? acc11 : acc10) : (nt ? acc01 : acc00);
      const int d  = nt * 16 + (lane & 15);
      const int hd = h * HDIM + d;
      const size_t row0 = (size_t)r_i * SEQ + i0 + ti * 16 + mb;
#pragma unroll
      for (int vv = 0; vv < 8; ++vv) {
        const size_t off = (row0 + vv) * HDTOT + hd;
        const float g = (float)gate[off];
        ctx[off] = (_Float16)(g * acc[vv]);
      }
    }
  }
}

// ---------------------------------------------------------------------------
// K5b: out = ctx[98304x256] @ w_out[256x128] + b_out  (fp32 out)
// One wave per 32x32 tile (2x2 blocking), K=256 -> 8 steps x 4 WMMA.
// ---------------------------------------------------------------------------
__global__ void __launch_bounds__(128) k_outproj(
    const _Float16* __restrict__ ctx, const _Float16* __restrict__ woutT,
    const float* __restrict__ b_out, float* __restrict__ out) {
  const int lane = threadIdx.x & 31, wave = threadIdx.x >> 5;
  const int gw   = blockIdx.x * 4 + wave;           // 3072 * 4 = 12288 waves
  const int mt2  = gw >> 2, nt2 = gw & 3;
  const int row0 = mt2 * 32, n0 = nt2 * 32;
  const _Float16* a0p = ctx + (size_t)row0 * HDTOT;
  const _Float16* a1p = a0p + (size_t)16 * HDTOT;
  const _Float16* b0p = woutT + (size_t)n0 * HDTOT;
  const _Float16* b1p = b0p + (size_t)16 * HDTOT;
  v8f acc00 = {}, acc01 = {}, acc10 = {}, acc11 = {};
#pragma unroll
  for (int s = 0; s < HDTOT / 32; ++s) {
    const v16h a0 = load_a(a0p + s * 32, HDTOT, lane);
    const v16h a1 = load_a(a1p + s * 32, HDTOT, lane);
    const v16h b0 = load_b_nmajor(b0p + s * 32, HDTOT, lane);
    const v16h b1 = load_b_nmajor(b1p + s * 32, HDTOT, lane);
    acc00 = wmma16x16x32(a0, b0, acc00);
    acc01 = wmma16x16x32(a0, b1, acc01);
    acc10 = wmma16x16x32(a1, b0, acc10);
    acc11 = wmma16x16x32(a1, b1, acc11);
  }
  const int mb = (lane >> 4) * 8;
#pragma unroll
  for (int ti = 0; ti < 2; ++ti) {
#pragma unroll
    for (int tj = 0; tj < 2; ++tj) {
      const v8f acc = ti ? (tj ? acc11 : acc10) : (tj ? acc01 : acc00);
      const int n = n0 + tj * 16 + (lane & 15);
      const float bo = b_out[n];
#pragma unroll
      for (int vv = 0; vv < 8; ++vv)
        out[(size_t)(row0 + ti * 16 + mb + vv) * IN_DIMC + n] = acc[vv] + bo;
    }
  }
}

// ---------------------------------------------------------------------------
extern "C" void kernel_launch(void* const* d_in, const int* in_sizes, int n_in,
                              void* d_out, int out_size, void* d_ws, size_t ws_size,
                              hipStream_t stream) {
  const float* m      = (const float*)d_in[0];
  const float* z      = (const float*)d_in[1];
  const float* ln_w   = (const float*)d_in[2];
  const float* ln_b   = (const float*)d_in[3];
  const float* w_qkv  = (const float*)d_in[4];
  const float* w_pair = (const float*)d_in[5];
  const float* w_gate = (const float*)d_in[6];
  const float* b_gate = (const float*)d_in[7];
  const float* w_out  = (const float*)d_in[8];
  const float* b_out  = (const float*)d_in[9];
  float* out = (float*)d_out;

  char* ws = (char*)d_ws;
  size_t off = 0;
  auto alloc = [&](size_t bytes) { char* p = ws + off; off += (bytes + 255) & ~(size_t)255; return p; };
  int*      probe  = (int*)alloc(256);
  _Float16* wqgT   = (_Float16*)alloc((size_t)NTOT * IN_DIMC * 2);          // 256 KB
  _Float16* woutT  = (_Float16*)alloc((size_t)IN_DIMC * HDTOT * 2);         // 64 KB
  _Float16* qb     = (_Float16*)alloc((size_t)NHEAD * SEQ * RDK * 2);       // 50.3 MB
  _Float16* kb     = (_Float16*)alloc((size_t)NHEAD * SEQ * RDK * 2);       // 50.3 MB
  _Float16* vTb    = (_Float16*)alloc((size_t)NHEAD * NROWM * HDIM * SEQ * 2); // 50.3 MB
  _Float16* gateb  = (_Float16*)alloc((size_t)ROWS * HDTOT * 2);            // 50.3 MB
  float*    biasb  = (float*)alloc((size_t)NHEAD * SEQ * SEQ * 4);          // 4.7 MB
  float*    logitb = (float*)alloc((size_t)NHEAD * SEQ * SEQ * 4);          // 4.7 MB
  _Float16* attnb  = (_Float16*)alloc((size_t)NHEAD * SEQ * SEQ * 2);       // 2.4 MB
  _Float16* ctxb   = (_Float16*)alloc((size_t)ROWS * HDTOT * 2);            // 50.3 MB
  (void)ws_size; (void)in_sizes; (void)n_in; (void)out_size;

  k_cdna5_probe<<<dim3(1), dim3(32), 0, stream>>>(probe);
  k_cvt_weights<<<dim3((NTOT * IN_DIMC + IN_DIMC * HDTOT + 255) / 256), dim3(256), 0, stream>>>(
      w_qkv, w_gate, w_out, wqgT, woutT);
  k_ln_project<<<dim3(ROWS / 16), dim3(256), 0, stream>>>(
      m, ln_w, ln_b, b_gate, wqgT, qb, kb, vTb, gateb);
  k_pair_bias<<<dim3((SEQ * SEQ / 16) / 4), dim3(128), 0, stream>>>(z, w_pair, biasb);
  k_logits<<<dim3(NHEAD * 12 * 12 / 4), dim3(128), 0, stream>>>(qb, kb, biasb, logitb);
  k_softmax<<<dim3(NHEAD * SEQ / 8), dim3(256), 0, stream>>>(logitb, attnb);
  k_context<<<dim3(NHEAD * NROWM * 12 / 4), dim3(128), 0, stream>>>(attnb, vTb, gateb, ctxb);
  k_outproj<<<dim3((ROWS / 32) * (IN_DIMC / 32) / 4), dim3(128), 0, stream>>>(
      ctxb, woutT, b_out, out);
}
